// bandwise_FF_69441031242164
// MI455X (gfx1250) — matrix operand order repaired and checked by reading, not compiled, not run on hardware
//
#include <hip/hip_runtime.h>
#include <hip/hip_fp16.h>

typedef __attribute__((ext_vector_type(16))) _Float16 v16h;
typedef __attribute__((ext_vector_type(8)))  _Float16 v8h;
typedef __attribute__((ext_vector_type(8)))  float    v8f;

#define NBANDS 21
#define CONDN  128
#define TOTALW 100
#define BATCHN 65536
#define MTILE  64
#define SUBS   8
#define ROWS_PER_WG (MTILE * SUBS)   /* 512 */
#define K1MAX  64                    /* layer-1 K padded (max 2w = 44) */
#define N3MAX  32                    /* layer-3 N padded (max w = 22)  */

__device__ __constant__ int c_lo[NBANDS] = {0,1,2,3,4,5,6,7,8,10,12,14,16,20,24,28,34,40,48,60,78};
__device__ __constant__ int c_w [NBANDS] = {1,1,1,1,1,1,1,1,2, 2, 2, 2, 4, 4, 4, 6, 6, 8,12,18,22};

struct BandPtrs {
    const float* W1[NBANDS]; const float* b1[NBANDS];
    const float* W2[NBANDS]; const float* b2[NBANDS];
    const float* W3[NBANDS]; const float* b3[NBANDS];
};

// Hardware tanh (v_tanh_f32 on gfx1250), branch-free fallback otherwise.
__device__ __forceinline__ float fast_tanh(float x) {
#if __has_builtin(__builtin_amdgcn_tanhf)
    return __builtin_amdgcn_tanhf(x);
#else
    float t = __builtin_amdgcn_exp2f(x * 2.8853900817779268f);  // exp(2x)
    return 1.0f - 2.0f * __builtin_amdgcn_rcpf(t + 1.0f);
#endif
}

// A-matrix fragment (16x32 f16), ISA 7.12.2: lane m = lane&15,
// halves 0..7 -> K = kb..kb+7, halves 8..15 -> K = kb+16..kb+23, kb = (lane>=16)*8
__device__ __forceinline__ v16h load_fragA(const _Float16* base, int row0, int ld,
                                           int k0, int lane) {
    int r  = row0 + (lane & 15);
    int kb = k0 + ((lane >> 4) << 3);
    const _Float16* p = base + r * ld + kb;
    v8h a0 = *(const v8h*)(p);
    v8h a1 = *(const v8h*)(p + 16);
    v16h a;
#pragma unroll
    for (int i = 0; i < 8; ++i) { a[i] = a0[i]; a[i + 8] = a1[i]; }
    return a;
}

// B-matrix fragment (32x16 f16), stored transposed in LDS as [N][K]:
// lane n = lane&15, halves 0..7 -> K = kb..kb+7, halves 8..15 -> K = kb+16..kb+23, kb = (lane>=16)*8
__device__ __forceinline__ v16h load_fragB(const _Float16* base, int n0, int ld,
                                           int k0, int lane) {
    int n  = n0 + (lane & 15);
    int kb = k0 + ((lane >> 4) << 3);
    const _Float16* p = base + n * ld + kb;
    v8h b0 = *(const v8h*)(p);
    v8h b1 = *(const v8h*)(p + 16);
    v16h b;
#pragma unroll
    for (int i = 0; i < 8; ++i) { b[i] = b0[i]; b[i + 8] = b1[i]; }
    return b;
}

__global__ __launch_bounds__(256)
void bandwise_ff_wmma_kernel(const float* __restrict__ x, float* __restrict__ out,
                             BandPtrs P) {
    __shared__ __align__(16) _Float16 sW1T[CONDN][K1MAX];   // 16 KB  [n][k]
    __shared__ __align__(16) _Float16 sW2T[CONDN][CONDN];   // 32 KB  [n][k]
    __shared__ __align__(16) _Float16 sW3T[N3MAX][CONDN];   //  8 KB  [n][k]
    __shared__ __align__(16) _Float16 sX [MTILE][K1MAX];    //  8 KB  [m][k]
    __shared__ __align__(16) _Float16 sH1[MTILE][CONDN];    // 16 KB
    __shared__ __align__(16) _Float16 sH2[MTILE][CONDN];    // 16 KB
    __shared__ float sB1[CONDN], sB2[CONDN], sB3[N3MAX];

    const int band = blockIdx.y;
    const int lo   = c_lo[band];
    const int w    = c_w[band];
    const int K1   = 2 * w;
    const bool twoK  = (K1 > 32);        // layer-1 needs second k-step (bands 19,20)
    const bool twoN3 = (w > 16);         // layer-3 needs second n-tile (bands 19,20)
    const int K1pad  = twoK  ? 64 : 32;
    const int kshift = twoK  ? 6  : 5;
    const int N3pad  = twoN3 ? 32 : 16;
    const int tid  = threadIdx.x;
    const int lane = tid & 31;
    const int wv   = tid >> 5;

    const float* gW1 = P.W1[band]; const float* gb1 = P.b1[band];
    const float* gW2 = P.W2[band]; const float* gb2 = P.b2[band];
    const float* gW3 = P.W3[band]; const float* gb3 = P.b3[band];

    // ---- stage weights into LDS (f16, transposed to [N][K], zero-padded) ----
    for (int i = tid; i < CONDN * K1pad; i += 256) {
        int n = i >> kshift, k = i & (K1pad - 1);
        sW1T[n][k] = (_Float16)((k < K1) ? gW1[k * CONDN + n] : 0.0f);
    }
    for (int i = tid; i < CONDN * CONDN; i += 256) {
        int n = i >> 7, k = i & (CONDN - 1);
        sW2T[n][k] = (_Float16)gW2[k * CONDN + n];
    }
    for (int i = tid; i < N3pad * CONDN; i += 256) {
        int n = i >> 7, k = i & (CONDN - 1);
        sW3T[n][k] = (_Float16)((n < w) ? gW3[k * w + n] : 0.0f);
    }
    if (tid < CONDN) { sB1[tid] = gb1[tid]; sB2[tid] = gb2[tid]; }
    if (tid < N3MAX) { sB3[tid] = (tid < w) ? gb3[tid] : 0.0f; }
    __syncthreads();

    // ---- stream 8 sub-tiles of 64 batch rows through the 3-layer MLP ----
    for (int s = 0; s < SUBS; ++s) {
        const int m0 = (blockIdx.x * SUBS + s) * MTILE;

        // gather band input slice -> f16, zero-pad K to K1pad
        for (int i = tid; i < MTILE * K1pad; i += 256) {
            int r = i >> kshift, k = i & (K1pad - 1);
            float v = 0.0f;
            if (k < K1) {
                int col = (k < w) ? (lo + k) : (TOTALW + lo + (k - w));
                v = x[(size_t)(m0 + r) * (2 * TOTALW) + col];
            }
            sX[r][k] = (_Float16)v;
        }
        __syncthreads();

        // ---- layer 1: (64 x K1pad) @ (K1pad x 128), tanh ----
        {
            const int n0 = wv * 16;
            const int n  = n0 + (lane & 15);
            const float bia = sB1[n];
            v16h b0 = load_fragB(&sW1T[0][0], n0, K1MAX, 0, lane);
            v16h b1;
            if (twoK) b1 = load_fragB(&sW1T[0][0], n0, K1MAX, 32, lane);
#pragma unroll
            for (int mt = 0; mt < 4; ++mt) {
                v8f acc = {};
                v16h a0 = load_fragA(&sX[0][0], mt * 16, K1MAX, 0, lane);
                acc = __builtin_amdgcn_wmma_f32_16x16x32_f16(
                    false, a0, false, b0, (short)0, acc, false, false);
                if (twoK) {   // block-uniform: scalar branch, EXEC stays all-ones
                    v16h a1 = load_fragA(&sX[0][0], mt * 16, K1MAX, 32, lane);
                    acc = __builtin_amdgcn_wmma_f32_16x16x32_f16(
                        false, a1, false, b1, (short)0, acc, false, false);
                }
                int mr = mt * 16 + ((lane >> 4) << 3);
#pragma unroll
                for (int e = 0; e < 8; ++e)
                    sH1[mr + e][n] = (_Float16)fast_tanh(acc[e] + bia);
            }
        }
        __syncthreads();

        // ---- layer 2: (64 x 128) @ (128 x 128), tanh ----
        {
            const int n0 = wv * 16;
            const int n  = n0 + (lane & 15);
            const float bia = sB2[n];
            v16h b0 = load_fragB(&sW2T[0][0], n0, CONDN,  0, lane);
            v16h b1 = load_fragB(&sW2T[0][0], n0, CONDN, 32, lane);
            v16h b2 = load_fragB(&sW2T[0][0], n0, CONDN, 64, lane);
            v16h b3 = load_fragB(&sW2T[0][0], n0, CONDN, 96, lane);
#pragma unroll
            for (int mt = 0; mt < 4; ++mt) {
                v8f acc = {};
                v16h a0 = load_fragA(&sH1[0][0], mt * 16, CONDN,  0, lane);
                acc = __builtin_amdgcn_wmma_f32_16x16x32_f16(
                    false, a0, false, b0, (short)0, acc, false, false);
                v16h a1 = load_fragA(&sH1[0][0], mt * 16, CONDN, 32, lane);
                acc = __builtin_amdgcn_wmma_f32_16x16x32_f16(
                    false, a1, false, b1, (short)0, acc, false, false);
                v16h a2 = load_fragA(&sH1[0][0], mt * 16, CONDN, 64, lane);
                acc = __builtin_amdgcn_wmma_f32_16x16x32_f16(
                    false, a2, false, b2, (short)0, acc, false, false);
                v16h a3 = load_fragA(&sH1[0][0], mt * 16, CONDN, 96, lane);
                acc = __builtin_amdgcn_wmma_f32_16x16x32_f16(
                    false, a3, false, b3, (short)0, acc, false, false);
                int mr = mt * 16 + ((lane >> 4) << 3);
#pragma unroll
                for (int e = 0; e < 8; ++e)
                    sH2[mr + e][n] = (_Float16)fast_tanh(acc[e] + bia);
            }
        }
        __syncthreads();

        // ---- layer 3: (64 x 128) @ (128 x N3pad), write valid cols ----
        {
            const int mt = wv & 3;
            const int nt = wv >> 2;
            // wave-uniform skip of the all-padding second n-tile
            if (nt == 0 || twoN3) {
                v8f acc = {};
#pragma unroll
                for (int kt = 0; kt < 4; ++kt) {
                    v16h a = load_fragA(&sH2[0][0],  mt * 16, CONDN, kt * 32, lane);
                    v16h b = load_fragB(&sW3T[0][0], nt * 16, CONDN, kt * 32, lane);
                    acc = __builtin_amdgcn_wmma_f32_16x16x32_f16(
                        false, a, false, b, (short)0, acc, false, false);
                }
                const int n = nt * 16 + (lane & 15);
                if (n < w) {
                    const float bia = sB3[n];
                    const int mr = m0 + mt * 16 + ((lane >> 4) << 3);
#pragma unroll
                    for (int e = 0; e < 8; ++e)
                        out[(size_t)(mr + e) * TOTALW + lo + n] = acc[e] + bia;
                }
            }
        }
        __syncthreads();
    }
}

extern "C" void kernel_launch(void* const* d_in, const int* in_sizes, int n_in,
                              void* d_out, int out_size, void* d_ws, size_t ws_size,
                              hipStream_t stream) {
    (void)in_sizes; (void)n_in; (void)out_size; (void)d_ws; (void)ws_size;
    const float* x = (const float*)d_in[0];
    BandPtrs P;
    for (int i = 0; i < NBANDS; ++i) {
        P.W1[i] = (const float*)d_in[1 + 0 * NBANDS + i];
        P.b1[i] = (const float*)d_in[1 + 1 * NBANDS + i];
        P.W2[i] = (const float*)d_in[1 + 2 * NBANDS + i];
        P.b2[i] = (const float*)d_in[1 + 3 * NBANDS + i];
        P.W3[i] = (const float*)d_in[1 + 4 * NBANDS + i];
        P.b3[i] = (const float*)d_in[1 + 5 * NBANDS + i];
    }
    dim3 grid(BATCHN / ROWS_PER_WG, NBANDS);
    bandwise_ff_wmma_kernel<<<grid, 256, 0, stream>>>(x, (float*)d_out, P);
}
